// RNN_50105088475509
// MI455X (gfx1250) — matrix operand order repaired and checked by reading, not compiled, not run on hardware
//
#include <hip/hip_runtime.h>
#include <hip/hip_bf16.h>
#include <math.h>

typedef __bf16 bf16_t;
typedef __attribute__((ext_vector_type(16))) __bf16 v16bf;
typedef __attribute__((ext_vector_type(8)))  __bf16 v8bf;
typedef __attribute__((ext_vector_type(8)))  float  v8f;

union BF16x16 { v16bf v; v8bf h2[2]; bf16_t e[16]; };
union F32x8   { v8f v; float e[8]; };

constexpr int kB   = 64;
constexpr int kT   = 2048;     // power of two (>> 11)
constexpr int kDin = 64;
constexpr int kH   = 128;
constexpr int kHS  = 136;      // padded LDS row stride (bf16 elems) -> bank spread
constexpr int kSlot = kB * kH; // 8192 floats per time slot in out[T+1][B][H]

__device__ __forceinline__ float fast_tanh(float x) {
#if __has_builtin(__builtin_amdgcn_tanhf)
  return __builtin_amdgcn_tanhf(x);
#elif __has_builtin(__builtin_amdgcn_tanh_f32)
  return __builtin_amdgcn_tanh_f32(x);
#else
  return tanhf(x);
#endif
}

// ---------------------------------------------------------------------------
// Phase 1: x_proj[b,t,:] = batch[b,t,:] @ W1^T + b1, written into out[t+1]
// One wave per 16-row M-tile (rows r = b*T + t), loops over 8 N-tiles.
// ---------------------------------------------------------------------------
__global__ __launch_bounds__(256) void rnn_proj_kernel(
    const float* __restrict__ batch, const float* __restrict__ W1,
    const float* __restrict__ b1, float* __restrict__ out) {
  const int wave  = blockIdx.x * 8 + (threadIdx.x >> 5);
  const int lane  = threadIdx.x & 31;
  const int half  = lane >> 4;   // 0: lanes 0-15, 1: lanes 16-31
  const int ln    = lane & 15;
  const int mbase = wave * 16;

  // A-matrix: 16 rows x K=64 bf16, contiguous K order k = 16*half + s (two 32-K chunks)
  BF16x16 A[2];
  const float* arow = batch + (size_t)(mbase + ln) * kDin;
#pragma unroll
  for (int c = 0; c < 2; ++c) {
    const int ko = c * 32 + half * 16;
#pragma unroll
    for (int j = 0; j < 8; ++j) A[c].e[j]     = (bf16_t)arow[ko + j];
#pragma unroll
    for (int j = 0; j < 8; ++j) A[c].e[8 + j] = (bf16_t)arow[ko + 8 + j];
  }

  for (int nt = 0; nt < 8; ++nt) {
    const int nbase = nt * 16;
    // B-matrix: B[k][n] = W1[n][k]; lane half selects K=0..15 / 16..31 of chunk
    const float* wrow = W1 + (size_t)(nbase + ln) * kDin;
    BF16x16 Bm[2];
#pragma unroll
    for (int c = 0; c < 2; ++c) {
      const int ko = c * 32 + half * 16;
#pragma unroll
      for (int j = 0; j < 16; ++j) Bm[c].e[j] = (bf16_t)wrow[ko + j];
    }
    const float bias = b1[nbase + ln];
    F32x8 acc;
#pragma unroll
    for (int j = 0; j < 8; ++j) acc.e[j] = bias;

    acc.v = __builtin_amdgcn_wmma_f32_16x16x32_bf16(false, A[0].v, false, Bm[0].v,
                                                    (short)0, acc.v, false, false);
    acc.v = __builtin_amdgcn_wmma_f32_16x16x32_bf16(false, A[1].v, false, Bm[1].v,
                                                    (short)0, acc.v, false, false);

    // C/D layout: VGPR j -> M = j + 8*half, N = ln. Row r = b*T + t.
#pragma unroll
    for (int j = 0; j < 8; ++j) {
      const int r = mbase + j + 8 * half;
      const int b = r >> 11;          // / kT
      const int t = r & (kT - 1);
      out[(size_t)(t + 1) * kSlot + b * kH + nbase + ln] = acc.e[j];
    }
  }
}

// ---------------------------------------------------------------------------
// Phase 2: sequential scan on ONE WGP. 32 waves = 4x8 grid of 16x16 tiles.
// h double-buffered in LDS (bf16) -> ONE barrier per step.
// xp_{t+1} prefetched into registers during step t -> L2 latency off the
// critical path. W2 tile held in registers for the whole 2048-step loop.
// out[t+1] is read as xp_t and overwritten in place with h_{t+1}.
// ---------------------------------------------------------------------------
__global__ __launch_bounds__(1024) void rnn_scan_kernel(
    const float* __restrict__ W2, const float* __restrict__ b2,
    float* __restrict__ out) {
  __shared__ __align__(16) bf16_t h_lds[2][kB * kHS];

  const int tid   = threadIdx.x;
  const int wave  = tid >> 5;
  const int lane  = tid & 31;
  const int half  = lane >> 4;
  const int ln    = lane & 15;
  const int mt    = wave >> 3;       // 0..3 (rows of h, batch dim)
  const int nt    = wave & 7;        // 0..7 (cols of h, hidden dim)
  const int mbase = mt * 16;
  const int nbase = nt * 16;

  // h0 = 0: zero both LDS state buffers and out slot 0
  for (int i = tid; i < kB * kHS; i += 1024) ((uint32_t*)h_lds)[i] = 0u; // 2 bufs * kB*kHS/2 dwords
  for (int i = tid; i < kSlot; i += 1024) out[i] = 0.0f;

  // Preload B tiles of W2^T as bf16 (held in VGPRs across all 2048 steps):
  // B[k][n] = W2[n][k]; 4 chunks of K=32.
  BF16x16 Bt[4];
  const float* wrow = W2 + (size_t)(nbase + ln) * kH;
#pragma unroll
  for (int c = 0; c < 4; ++c) {
    const int ko = c * 32 + half * 16;
#pragma unroll
    for (int j = 0; j < 16; ++j) Bt[c].e[j] = (bf16_t)wrow[ko + j];
  }
  const float bias = b2[nbase + ln];

  // Prefetch xp_0 from out slot 1 (lane-private elements; no intra-kernel race)
  F32x8 xp;
  {
    const float* s1 = out + kSlot;
#pragma unroll
    for (int j = 0; j < 8; ++j) {
      const int m = mbase + j + 8 * half;
      xp.e[j] = s1[m * kH + nbase + ln];
    }
  }

  __syncthreads();

  for (int t = 0; t < kT; ++t) {
    const int p = t & 1;
    float* slot = out + (size_t)(t + 1) * kSlot;

    // acc = xp_t + b2 (C/D layout), xp_t was prefetched last step
    F32x8 acc;
#pragma unroll
    for (int j = 0; j < 8; ++j) acc.e[j] = xp.e[j] + bias;

    // A = h tile (16 x 128 bf16) from LDS buffer p, contiguous K order k = 16*half + s, 4 K-chunks
    BF16x16 At[4];
#pragma unroll
    for (int c = 0; c < 4; ++c) {
      const bf16_t* hp = &h_lds[p][(mbase + ln) * kHS + c * 32 + half * 16];
      At[c].h2[0] = *(const v8bf*)hp;         // K = ko .. ko+7
      At[c].h2[1] = *(const v8bf*)(hp + 8);   // K = ko+8 .. ko+15
    }

    // Prefetch xp_{t+1} (clamped to last valid slot on the final iteration);
    // latency hidden behind WMMA + tanh + stores + barrier.
    {
      const float* nslot = out + (size_t)(t + 2 <= kT ? t + 2 : kT) * kSlot;
#pragma unroll
      for (int j = 0; j < 8; ++j) {
        const int m = mbase + j + 8 * half;
        xp.e[j] = nslot[m * kH + nbase + ln];
      }
    }

#pragma unroll
    for (int c = 0; c < 4; ++c)
      acc.v = __builtin_amdgcn_wmma_f32_16x16x32_bf16(false, At[c].v, false, Bt[c].v,
                                                      (short)0, acc.v, false, false);

    // h_new = tanh(acc): bf16 into the OTHER LDS buffer, f32 into out[t+1]
    bf16_t* hw = &h_lds[p ^ 1][0];
#pragma unroll
    for (int j = 0; j < 8; ++j) {
      const float hv = fast_tanh(acc.e[j]);
      const int m = mbase + j + 8 * half;
      hw[m * kHS + nbase + ln] = (bf16_t)hv;
      slot[m * kH + nbase + ln] = hv;
    }

    __syncthreads();  // single barrier per step: h(t+1) visible before step t+1
  }
}

extern "C" void kernel_launch(void* const* d_in, const int* in_sizes, int n_in,
                              void* d_out, int out_size, void* d_ws, size_t ws_size,
                              hipStream_t stream) {
  const float* batch = (const float*)d_in[0];
  const float* W1    = (const float*)d_in[1];
  const float* b1    = (const float*)d_in[2];
  const float* W2    = (const float*)d_in[3];
  const float* b2    = (const float*)d_in[4];
  float* out = (float*)d_out;

  // Phase 1: 131072 rows / 16 per wave / 8 waves per block = 1024 blocks
  rnn_proj_kernel<<<dim3((kB * kT) / (16 * 8)), dim3(256), 0, stream>>>(batch, W1, b1, out);
  // Phase 2: single workgroup (one WGP), 32 waves, sequential over T
  rnn_scan_kernel<<<dim3(1), dim3(1024), 0, stream>>>(W2, b2, out);
}
